// truncateWavenumbers_45045617000708
// MI455X (gfx1250) — matrix operand order repaired and checked by reading, not compiled, not run on hardware
//
#include <hip/hip_runtime.h>

// ---------------------------------------------------------------------------
// Spectral truncation as exact separable circulant GEMMs:
//   y = Ar * (P * x) + Ai * (Q * x)      (P,Q along W;  Ar,Ai along H)
// Bandwidth-bound problem (69 GFLOP vs ~268MB HBM floor).  Per-batch
// stage1/stage2 pipelining keeps the 16.8MB f16 intermediate in the 192MB L2.
// Matrix math runs on v_wmma_f32_16x16x32_f16 (wave32 CDNA5 WMMA).
// ---------------------------------------------------------------------------

typedef __attribute__((ext_vector_type(16))) _Float16 v16h;
typedef __attribute__((ext_vector_type(8)))  _Float16 v8h;
typedef __attribute__((ext_vector_type(4)))  _Float16 v4h;
typedef __attribute__((ext_vector_type(8)))  float    v8f;

#define HH 256
#define WW 256
#define CC 64
#define NB 8
#define LDSW 264   // padded LDS row stride in halves (528B = 33*16B, keeps 16B align)

union FragA { v16h v; v8h h[2]; };

// ---------------------------------------------------------------------------
// Basis precompute (runs every launch; trivially cheap, deterministic).
//   p(d)  = (1 + 2*sum_{k=1..31} cos(2πkd/256)) / 256
//   q(d)  = -(2/256) * sum_{k=1..31} sin(2πkd/256)
//   ar(d) = p(d) + cos(πd/4)/256
//   ai(d) = -sin(πd/4)/256
// ---------------------------------------------------------------------------
__global__ __launch_bounds__(256) void basis_kernel(
    _Float16* __restrict__ P, _Float16* __restrict__ Q,
    _Float16* __restrict__ Ar, _Float16* __restrict__ Ai) {
  int idx = blockIdx.x * blockDim.x + threadIdx.x;   // 0..65535 -> (i,j)
  int i = idx >> 8, j = idx & 255;
  float th = (6.283185307179586f / 256.0f) * (float)(i - j);
  float sumc = 0.0f, sums = 0.0f;
#pragma unroll 1
  for (int k = 1; k < 32; ++k) {
    sumc += cosf(th * (float)k);
    sums += sinf(th * (float)k);
  }
  float p  = (1.0f + 2.0f * sumc) * (1.0f / 256.0f);
  float q  = -2.0f * sums * (1.0f / 256.0f);
  float ar = p + cosf(th * 32.0f) * (1.0f / 256.0f);
  float ai = -sinf(th * 32.0f) * (1.0f / 256.0f);
  P[idx]  = (_Float16)p;
  Q[idx]  = (_Float16)q;
  Ar[idx] = (_Float16)ar;
  Ai[idx] = (_Float16)ai;
}

// ---------------------------------------------------------------------------
// Stage 1: per (b,h) slab, contract along W.
//   T1[h,w',c] = sum_w P[w',w] x[b,h,w,c] ;  T2 likewise with Q.
// GEMM M=256(w'), N=64(c), K=256(w). One block per h, 8 waves.
// ---------------------------------------------------------------------------
__global__ __launch_bounds__(256) void stage1_kernel(
    const float* __restrict__ x, const _Float16* __restrict__ P,
    const _Float16* __restrict__ Q, _Float16* __restrict__ T1,
    _Float16* __restrict__ T2, int b) {
  __shared__ _Float16 Xl[CC * LDSW];            // x slab transposed: [c][w], f16
  const int tid = threadIdx.x;
  const int h = blockIdx.x;
  const float* xs = x + ((size_t)b * HH + h) * (size_t)(WW * CC);  // [w][c]

  // Load [256 w][64 c] f32, convert f16, store transposed into LDS.
#pragma unroll
  for (int pass = 0; pass < 16; ++pass) {
    int w  = pass * 16 + (tid >> 4);
    int c0 = (tid & 15) * 4;
    float4 v = *(const float4*)(xs + (size_t)w * CC + c0);
    Xl[(c0 + 0) * LDSW + w] = (_Float16)v.x;
    Xl[(c0 + 1) * LDSW + w] = (_Float16)v.y;
    Xl[(c0 + 2) * LDSW + w] = (_Float16)v.z;
    Xl[(c0 + 3) * LDSW + w] = (_Float16)v.w;
  }
  __syncthreads();

  const int lane = tid & 31, wave = tid >> 5;
  const int hs = lane >> 4;       // lane half (0: lanes 0-15, 1: lanes 16-31)
  const int nr = lane & 15;       // A-row (M) / B-col (N) index

#pragma unroll
  for (int mi = 0; mi < 2; ++mi) {
    const int mTile = wave + mi * 8;            // 16 M-tiles over 8 waves x 2
    const _Float16* Prow = P + (size_t)(mTile * 16 + nr) * WW;
    const _Float16* Qrow = Q + (size_t)(mTile * 16 + nr) * WW;
    v8f accP[4] = {};
    v8f accQ[4] = {};
#pragma unroll
    for (int k = 0; k < 8; ++k) {
      const int kb = k * 32;
      FragA aP, aQ;
      // 16-bit A 16x32 layout: elem e<8 -> K=e+8*hs ; e>=8 -> K=e+8+8*hs
      aP.h[0] = *(const v8h*)(Prow + kb + 8 * hs);
      aP.h[1] = *(const v8h*)(Prow + kb + 16 + 8 * hs);
      aQ.h[0] = *(const v8h*)(Qrow + kb + 8 * hs);
      aQ.h[1] = *(const v8h*)(Qrow + kb + 16 + 8 * hs);
#pragma unroll
      for (int ct = 0; ct < 4; ++ct) {
        FragA bf;  // 16-bit B 32x16 layout: elem e<8 -> K=e+8*hs ; e>=8 -> K=e+8+8*hs, N = nr
        const _Float16* bp = &Xl[(ct * 16 + nr) * LDSW + kb + 8 * hs];
        bf.h[0] = *(const v8h*)(bp);
        bf.h[1] = *(const v8h*)(bp + 16);
        accP[ct] = __builtin_amdgcn_wmma_f32_16x16x32_f16(
            false, aP.v, false, bf.v, (short)0, accP[ct], false, false);
        accQ[ct] = __builtin_amdgcn_wmma_f32_16x16x32_f16(
            false, aQ.v, false, bf.v, (short)0, accQ[ct], false, false);
      }
    }
    // C/D layout: VGPR r holds M = r + 8*hs, N = nr.  Store T[h][w'][c] f16.
    const size_t baseT = (size_t)h * (WW * CC);
#pragma unroll
    for (int ct = 0; ct < 4; ++ct) {
#pragma unroll
      for (int r = 0; r < 8; ++r) {
        const int wp = mTile * 16 + r + 8 * hs;
        const int c  = ct * 16 + nr;
        T1[baseT + (size_t)wp * CC + c] = (_Float16)accP[ct][r];
        T2[baseT + (size_t)wp * CC + c] = (_Float16)accQ[ct][r];
      }
    }
  }
}

// ---------------------------------------------------------------------------
// Stage 2: per (b,w) slab, contract along H and combine:
//   y[b,h',w,c] = sum_h Ar[h',h] T1[h,w,c] + Ai[h',h] T2[h,w,c]
// ---------------------------------------------------------------------------
__global__ __launch_bounds__(256) void stage2_kernel(
    const _Float16* __restrict__ T1, const _Float16* __restrict__ T2,
    const _Float16* __restrict__ Ar, const _Float16* __restrict__ Ai,
    float* __restrict__ out, int b) {
  __shared__ _Float16 L1[CC * LDSW];            // T1 slab transposed [c][h]
  __shared__ _Float16 L2[CC * LDSW];            // T2 slab transposed [c][h]
  const int tid = threadIdx.x;
  const int w = blockIdx.x;

#pragma unroll
  for (int pass = 0; pass < 16; ++pass) {
    int h  = pass * 16 + (tid >> 4);
    int c0 = (tid & 15) * 4;
    v4h a = *(const v4h*)(T1 + ((size_t)h * WW + w) * CC + c0);
    v4h q = *(const v4h*)(T2 + ((size_t)h * WW + w) * CC + c0);
#pragma unroll
    for (int j = 0; j < 4; ++j) {
      L1[(c0 + j) * LDSW + h] = a[j];
      L2[(c0 + j) * LDSW + h] = q[j];
    }
  }
  __syncthreads();

  const int lane = tid & 31, wave = tid >> 5;
  const int hs = lane >> 4;
  const int nr = lane & 15;

#pragma unroll
  for (int mi = 0; mi < 2; ++mi) {
    const int mTile = wave + mi * 8;
    const _Float16* ArRow = Ar + (size_t)(mTile * 16 + nr) * HH;
    const _Float16* AiRow = Ai + (size_t)(mTile * 16 + nr) * HH;
    v8f acc[4] = {};
#pragma unroll
    for (int k = 0; k < 8; ++k) {
      const int kb = k * 32;
      FragA fr, fi;
      fr.h[0] = *(const v8h*)(ArRow + kb + 8 * hs);
      fr.h[1] = *(const v8h*)(ArRow + kb + 16 + 8 * hs);
      fi.h[0] = *(const v8h*)(AiRow + kb + 8 * hs);
      fi.h[1] = *(const v8h*)(AiRow + kb + 16 + 8 * hs);
#pragma unroll
      for (int ct = 0; ct < 4; ++ct) {
        FragA b1, b2;
        const _Float16* p1 = &L1[(ct * 16 + nr) * LDSW + kb + 8 * hs];
        const _Float16* p2 = &L2[(ct * 16 + nr) * LDSW + kb + 8 * hs];
        b1.h[0] = *(const v8h*)(p1); b1.h[1] = *(const v8h*)(p1 + 16);
        b2.h[0] = *(const v8h*)(p2); b2.h[1] = *(const v8h*)(p2 + 16);
        acc[ct] = __builtin_amdgcn_wmma_f32_16x16x32_f16(
            false, fr.v, false, b1.v, (short)0, acc[ct], false, false);
        acc[ct] = __builtin_amdgcn_wmma_f32_16x16x32_f16(
            false, fi.v, false, b2.v, (short)0, acc[ct], false, false);
      }
    }
    // Store f32 output y[b][h'][w][c]; lanes contiguous over c.
#pragma unroll
    for (int ct = 0; ct < 4; ++ct) {
#pragma unroll
      for (int r = 0; r < 8; ++r) {
        const int hp = mTile * 16 + r + 8 * hs;
        const int c  = ct * 16 + nr;
        out[(((size_t)b * HH + hp) * WW + w) * CC + c] = acc[ct][r];
      }
    }
  }
}

// ---------------------------------------------------------------------------
extern "C" void kernel_launch(void* const* d_in, const int* in_sizes, int n_in,
                              void* d_out, int out_size, void* d_ws, size_t ws_size,
                              hipStream_t stream) {
  (void)in_sizes; (void)n_in; (void)out_size; (void)ws_size;
  const float* x = (const float*)d_in[0];
  float* out = (float*)d_out;

  // Workspace layout (halves): P, Q, Ar, Ai (64K each) then T1, T2 (4.19M each)
  _Float16* P  = (_Float16*)d_ws;
  _Float16* Q  = P  + (size_t)HH * WW;
  _Float16* Ar = Q  + (size_t)HH * WW;
  _Float16* Ai = Ar + (size_t)HH * WW;
  _Float16* T1 = Ai + (size_t)HH * WW;
  _Float16* T2 = T1 + (size_t)HH * WW * CC;

  basis_kernel<<<256, 256, 0, stream>>>(P, Q, Ar, Ai);

  // Per-batch pipelining: the 16.8MB T1/T2 intermediate stays in the 192MB L2
  // between stage1(b) and stage2(b), so HBM traffic ~= input + output only.
  for (int b = 0; b < NB; ++b) {
    stage1_kernel<<<HH, 256, 0, stream>>>(x, P, Q, T1, T2, b);
    stage2_kernel<<<WW, 256, 0, stream>>>(T1, T2, Ar, Ai, out, b);
  }
}
